// SelfAttentionBlock_57019985821758
// MI455X (gfx1250) — matrix operand order repaired and checked by reading, not compiled, not run on hardware
//
#include <hip/hip_runtime.h>
#include <hip/hip_bf16.h>
#include <stdint.h>

// Problem constants (from reference setup_inputs)
#define NB     2
#define CCH    64            // channels C
#define ACH    32            // attention channels A
#define DD     28
#define S_TOT  (DD*DD*DD)    // 21952 spatial sites
#define T_REAL (S_TOT/8)     // 2744 pooled sites
#define T_PAD  2752          // padded to multiple of 32 for 32-key chunks
#define ROWBLK (S_TOT/16)    // 1372 row blocks of 16
#define SBLK   ((S_TOT+255)/256)   // 86 s-blocks for BN partials
#define NEG_BIG (-3.0e38f)
#define BN_EPS 1e-5f

typedef __attribute__((ext_vector_type(16))) __bf16 v16bf;
typedef __attribute__((ext_vector_type(8)))  float  v8f;

union BF16Frag { v16bf v; uint32_t u[8]; };

__device__ __forceinline__ uint32_t f32_to_bf16_bits(float f) {
  union { __bf16 h; unsigned short s; } cv;
  cv.h = (__bf16)f;
  return (uint32_t)cv.s;
}

// ---------------------------------------------------------------------------
// Prep: thetaB[n][s][c] = bf16(x[n][c][s])   (row-major rows, c contiguous)
// ---------------------------------------------------------------------------
__global__ void k_theta(const float* __restrict__ x, __bf16* __restrict__ thetaB) {
  int idx = blockIdx.x * 256 + threadIdx.x;       // over N*S
  if (idx >= NB * S_TOT) return;
  int n = idx / S_TOT, s = idx % S_TOT;
  const float* xs = x + (size_t)n * CCH * S_TOT + s;
  __bf16* o = thetaB + (size_t)idx * CCH;
  #pragma unroll 8
  for (int c = 0; c < CCH; ++c) o[c] = (__bf16)xs[(size_t)c * S_TOT];
}

// ---------------------------------------------------------------------------
// Prep: phiT[n][t][c] = bf16(maxpool2x2x2(x)[n][c][t]); pad rows t>=2744 -> 0
// ---------------------------------------------------------------------------
__global__ void k_phi(const float* __restrict__ x, __bf16* __restrict__ phiT) {
  int idx = blockIdx.x * 256 + threadIdx.x;       // over N*T_PAD
  if (idx >= NB * T_PAD) return;
  int n = idx / T_PAD, t = idx % T_PAD;
  __bf16* o = phiT + (size_t)idx * CCH;
  if (t >= T_REAL) {
    #pragma unroll 8
    for (int c = 0; c < CCH; ++c) o[c] = (__bf16)0.0f;
    return;
  }
  int d2 = t / 196, r = t % 196, w2 = r / 14, h2 = r % 14;
  int base = d2 * 2 * 784 + w2 * 2 * 28 + h2 * 2;
  const float* xn = x + (size_t)n * CCH * S_TOT + base;
  #pragma unroll 4
  for (int c = 0; c < CCH; ++c) {
    const float* xc = xn + (size_t)c * S_TOT;
    float m = xc[0];
    m = fmaxf(m, xc[1]);   m = fmaxf(m, xc[28]);  m = fmaxf(m, xc[29]);
    m = fmaxf(m, xc[784]); m = fmaxf(m, xc[785]); m = fmaxf(m, xc[812]);
    m = fmaxf(m, xc[813]);
    o[c] = (__bf16)m;
  }
}

// ---------------------------------------------------------------------------
// Prep: gxT[n][a][t] = bf16( max_{8 children}(g_w[a]·x[:,child]) + g_b[a] )
// (bias constant per a, commutes with max). Pad t -> 0.
// ---------------------------------------------------------------------------
__global__ void k_g(const float* __restrict__ x, const float* __restrict__ g_w,
                    const float* __restrict__ g_b, __bf16* __restrict__ gxT) {
  int t = blockIdx.x * 256 + threadIdx.x;
  if (t >= T_PAD) return;
  int a = blockIdx.y, n = blockIdx.z;
  __bf16* o = gxT + ((size_t)n * ACH + a) * T_PAD + t;
  if (t >= T_REAL) { *o = (__bf16)0.0f; return; }
  int d2 = t / 196, r = t % 196, w2 = r / 14, h2 = r % 14;
  int base = d2 * 2 * 784 + w2 * 2 * 28 + h2 * 2;
  const float* xn = x + (size_t)n * CCH * S_TOT + base;
  const float* gw = g_w + (size_t)a * CCH;   // uniform per block -> scalar loads
  float d0=0,d1=0,d2s=0,d3=0,d4=0,d5=0,d6=0,d7=0;
  #pragma unroll 4
  for (int c = 0; c < CCH; ++c) {
    const float* xc = xn + (size_t)c * S_TOT;
    float w = gw[c];
    d0  = fmaf(w, xc[0],   d0);  d1 = fmaf(w, xc[1],   d1);
    d2s = fmaf(w, xc[28],  d2s); d3 = fmaf(w, xc[29],  d3);
    d4  = fmaf(w, xc[784], d4);  d5 = fmaf(w, xc[785], d5);
    d6  = fmaf(w, xc[812], d6);  d7 = fmaf(w, xc[813], d7);
  }
  float m = fmaxf(fmaxf(fmaxf(d0, d1), fmaxf(d2s, d3)),
                  fmaxf(fmaxf(d4, d5), fmaxf(d6, d7)));
  *o = (__bf16)(m + g_b[a]);
}

// ---------------------------------------------------------------------------
// Flash attention core, TRANSPOSED tiling. One wave = 16 query rows.
// Everything is computed as the transpose so the key dimension lies within
// each lane and the P-probability tile lands directly in B-fragment layout:
//   S^T[key][row] = phi_key · theta_row :
//       A = phi chunk (2 frags of 16 keys), B = theta^T (loop-invariant)
//       4x v_wmma_f32_16x16x32_bf16
//   softmax over keys: 15 in-lane VALU + ONE half-swap shuffle (max and sum)
//   y^T[a][row] += V^T · P^T :
//       A = gxT (2 frags of 16 a-cols), B = P^T built with 4 half-swap
//       shuffles + selects (no LDS at all)
//       2x v_wmma_f32_16x16x32_bf16
// ---------------------------------------------------------------------------
__global__ void __launch_bounds__(128, 1)
k_attn(const __bf16* __restrict__ thetaB, const __bf16* __restrict__ phiT,
       const __bf16* __restrict__ gxT, float* __restrict__ y) {
  const int lane = threadIdx.x & 31;
  const int half = lane >> 4;
  const int lo   = lane & 15;
  const int wid  = blockIdx.x * 4 + (threadIdx.x >> 5);  // 0..NB*ROWBLK-1
  const int n    = wid / ROWBLK;
  const int row0 = (wid % ROWBLK) * 16;

  // theta^T as B matrix [K=64c x N=16rows]; B layout: N=lo, K=2v+16*half.
  // Loop-invariant: two c-halves.
  BF16Frag tb0, tb1;
  {
    const uint32_t* tr =
        (const uint32_t*)(thetaB + ((size_t)n * S_TOT + row0 + lo) * CCH);
    #pragma unroll
    for (int v = 0; v < 8; ++v) {
      int ci = v + 8 * half;
      tb0.u[v] = tr[ci];        // c 0..31
      tb1.u[v] = tr[16 + ci];   // c 32..63
    }
  }

  v8f acc0 = {}, acc1 = {};      // y^T frags: a = 16f + 8*half + v, col = lo
  float mrow = NEG_BIG, lrow = 0.0f;   // per-lane scalars (row = lo)

  const uint32_t* phi_base = (const uint32_t*)(phiT + (size_t)n * T_PAD * CCH);
  const uint32_t* gx_base  = (const uint32_t*)(gxT + (size_t)n * ACH * T_PAD);

  for (int t = 0; t < T_PAD / 32; ++t) {
    const int kb = t * 32;

    // phi as A matrices: frag f covers key = kb + 16f + lo.
    // A layout: K=2v+16*half
    BF16Frag pa00, pa01, pa10, pa11;   // [key-frag][c-half]
    {
      const uint32_t* p0 = phi_base + (size_t)(kb + lo) * (CCH / 2);
      const uint32_t* p1 = phi_base + (size_t)(kb + 16 + lo) * (CCH / 2);
      #pragma unroll
      for (int v = 0; v < 8; ++v) {
        int ki = v + 8 * half;
        pa00.u[v] = p0[ki];  pa01.u[v] = p0[16 + ki];
        pa10.u[v] = p1[ki];  pa11.u[v] = p1[16 + ki];
      }
    }

    // logits^T: s_f holds key m = kb + 16f + v + 8*half, row = lo
    v8f s0 = {}, s1 = {};
    s0 = __builtin_amdgcn_wmma_f32_16x16x32_bf16(false, pa00.v, false, tb0.v,
                                                 (short)0, s0, false, false);
    s0 = __builtin_amdgcn_wmma_f32_16x16x32_bf16(false, pa01.v, false, tb1.v,
                                                 (short)0, s0, false, false);
    s1 = __builtin_amdgcn_wmma_f32_16x16x32_bf16(false, pa10.v, false, tb0.v,
                                                 (short)0, s1, false, false);
    s1 = __builtin_amdgcn_wmma_f32_16x16x32_bf16(false, pa11.v, false, tb1.v,
                                                 (short)0, s1, false, false);

    // mask padded keys (statically only the last chunk)
    if (kb + 32 > T_REAL) {
      #pragma unroll
      for (int v = 0; v < 8; ++v) {
        if (kb + v + 8 * half >= T_REAL)      s0[v] = NEG_BIG;
        if (kb + 16 + v + 8 * half >= T_REAL) s1[v] = NEG_BIG;
      }
    }

    // chunk max over keys: in-lane tree + one half swap
    float cmx = fmaxf(s0[0], s1[0]);
    #pragma unroll
    for (int v = 1; v < 8; ++v) cmx = fmaxf(cmx, fmaxf(s0[v], s1[v]));
    cmx = fmaxf(cmx, __shfl_xor(cmx, 16, 32));
    float mnew = fmaxf(mrow, cmx);
    float alpha = __expf(mrow - mnew);
    mrow = mnew;

    // probabilities + row sum (one half-swap shuffle)
    float p0[8], p1[8], ls = 0.0f;
    #pragma unroll
    for (int v = 0; v < 8; ++v) {
      p0[v] = __expf(s0[v] - mnew);
      p1[v] = __expf(s1[v] - mnew);
      ls += p0[v] + p1[v];
    }
    ls += __shfl_xor(ls, 16, 32);
    lrow = lrow * alpha + ls;
    #pragma unroll
    for (int v = 0; v < 8; ++v) { acc0[v] *= alpha; acc1[v] *= alpha; }

    // Build P^T as a B fragment [K=32 keys x N=16 rows]:
    //   B must match va's k order: lane(half h, lo) holds keys 16h..16h+15 packed 2/VGPR.
    //   Locally we hold keys {8h..8h+7} (s0) and {16+8h..16+8h+7} (s1):
    //   pack pairs then swap the "wrong" quads across wave halves.
    BF16Frag pb;
    #pragma unroll
    for (int j = 0; j < 4; ++j) {
      uint32_t q0 = f32_to_bf16_bits(p0[2*j]) | (f32_to_bf16_bits(p0[2*j+1]) << 16);
      uint32_t q1 = f32_to_bf16_bits(p1[2*j]) | (f32_to_bf16_bits(p1[2*j+1]) << 16);
      uint32_t send = half ? q0 : q1;
      uint32_t recv = (uint32_t)__shfl_xor((int)send, 16, 32);
      pb.u[j]     = half ? recv : q0;   // keys 16h + 0..7
      pb.u[4 + j] = half ? q1 : recv;   // keys 16h + 8..15
    }

    // V^T as A matrices: frag f covers a = 16f + lo; K = keys in chunk.
    BF16Frag va0, va1;
    {
      const uint32_t* g0 = gx_base + (size_t)lo * (T_PAD / 2) + kb / 2;
      const uint32_t* g1 = gx_base + (size_t)(16 + lo) * (T_PAD / 2) + kb / 2;
      #pragma unroll
      for (int v = 0; v < 8; ++v) {
        int ki = v + 8 * half;
        va0.u[v] = g0[ki];
        va1.u[v] = g1[ki];
      }
    }
    acc0 = __builtin_amdgcn_wmma_f32_16x16x32_bf16(false, va0.v, false, pb.v,
                                                   (short)0, acc0, false, false);
    acc1 = __builtin_amdgcn_wmma_f32_16x16x32_bf16(false, va1.v, false, pb.v,
                                                   (short)0, acc1, false, false);
  }

  // epilogue: y[n][row][a] = acc^T / l ; per lane row = lo, a = 16f + 8h + v
  // (contiguous v -> compiler can emit b128 stores)
  float inv = 1.0f / lrow;
  float* yr = y + ((size_t)n * S_TOT + row0 + lo) * ACH + 8 * half;
  #pragma unroll
  for (int v = 0; v < 8; ++v) {
    yr[v]      = acc0[v] * inv;
    yr[16 + v] = acc1[v] * inv;
  }
}

// ---------------------------------------------------------------------------
// z[n][c][s] = sum_a y[n][s][a] * w_w[c][a] + w_b[c]; per-block BN partials.
// grid: (SBLK, C, N); deterministic two-pass stats (no float atomics).
// ---------------------------------------------------------------------------
__global__ void __launch_bounds__(256)
k_z(const float* __restrict__ y, const float* __restrict__ w_w,
    const float* __restrict__ w_b, float* __restrict__ z,
    float* __restrict__ partS, float* __restrict__ partQ) {
  __shared__ float redS[256];
  __shared__ float redQ[256];
  const int c = blockIdx.y, n = blockIdx.z;
  const int s = blockIdx.x * 256 + threadIdx.x;
  const float* wr = w_w + (size_t)c * ACH;     // uniform -> scalar loads
  float zv = 0.0f;
  if (s < S_TOT) {
    const float* yr = y + ((size_t)n * S_TOT + s) * ACH;
    float d = w_b[c];
    #pragma unroll
    for (int a = 0; a < ACH; ++a) d = fmaf(yr[a], wr[a], d);
    z[((size_t)n * CCH + c) * S_TOT + s] = d;
    zv = d;
  }
  redS[threadIdx.x] = zv;
  redQ[threadIdx.x] = zv * zv;
  __syncthreads();
  for (int off = 128; off > 0; off >>= 1) {
    if (threadIdx.x < off) {
      redS[threadIdx.x] += redS[threadIdx.x + off];
      redQ[threadIdx.x] += redQ[threadIdx.x + off];
    }
    __syncthreads();
  }
  if (threadIdx.x == 0) {
    size_t pi = ((size_t)n * CCH + c) * SBLK + blockIdx.x;
    partS[pi] = redS[0];
    partQ[pi] = redQ[0];
  }
}

// ---------------------------------------------------------------------------
// Finalize BN: per-channel scale/bias from partials (deterministic order).
// ---------------------------------------------------------------------------
__global__ void k_bn(const float* __restrict__ partS, const float* __restrict__ partQ,
                     const float* __restrict__ gamma, const float* __restrict__ beta,
                     float* __restrict__ sb) {
  int c = threadIdx.x;
  if (c >= CCH) return;
  float s = 0.0f, q = 0.0f;
  for (int n = 0; n < NB; ++n)
    for (int b = 0; b < SBLK; ++b) {
      size_t pi = ((size_t)n * CCH + c) * SBLK + b;
      s += partS[pi];
      q += partQ[pi];
    }
  const float cnt = (float)(NB * S_TOT);
  float mean = s / cnt;
  float var  = q / cnt - mean * mean;
  float sc   = gamma[c] * rsqrtf(var + BN_EPS);
  sb[c]        = sc;
  sb[CCH + c]  = beta[c] - mean * sc;
}

// ---------------------------------------------------------------------------
// out = z*scale[c] + bias[c] + x   (elementwise residual)
// ---------------------------------------------------------------------------
__global__ void k_out(const float* __restrict__ z, const float* __restrict__ sb,
                      const float* __restrict__ x, float* __restrict__ out) {
  size_t idx = (size_t)blockIdx.x * 256 + threadIdx.x;
  if (idx >= (size_t)NB * CCH * S_TOT) return;
  int c = (int)((idx / S_TOT) % CCH);
  out[idx] = fmaf(z[idx], sb[c], sb[CCH + c] + x[idx]);
}

// ---------------------------------------------------------------------------
extern "C" void kernel_launch(void* const* d_in, const int* in_sizes, int n_in,
                              void* d_out, int out_size, void* d_ws, size_t ws_size,
                              hipStream_t stream) {
  const float* x     = (const float*)d_in[0];
  const float* g_w   = (const float*)d_in[1];
  const float* g_b   = (const float*)d_in[2];
  const float* w_w   = (const float*)d_in[3];
  const float* w_b   = (const float*)d_in[4];
  const float* gamma = (const float*)d_in[5];
  const float* beta  = (const float*)d_in[6];
  float* out = (float*)d_out;
  (void)in_sizes; (void)n_in; (void)out_size; (void)ws_size;

  char* ws = (char*)d_ws;
  size_t off = 0;
  auto take = [&](size_t bytes) -> char* {
    char* p = ws + off;
    off += (bytes + 255) & ~(size_t)255;
    return p;
  };
  __bf16* thetaB = (__bf16*)take((size_t)NB * S_TOT * CCH * 2);   // 5.6 MB
  __bf16* phiT   = (__bf16*)take((size_t)NB * T_PAD * CCH * 2);   // 0.7 MB
  __bf16* gxT    = (__bf16*)take((size_t)NB * ACH * T_PAD * 2);   // 0.35 MB
  float*  y      = (float*)take((size_t)NB * S_TOT * ACH * 4);    // 5.6 MB
  float*  z      = (float*)take((size_t)NB * CCH * S_TOT * 4);    // 11.2 MB
  float*  partS  = (float*)take((size_t)NB * CCH * SBLK * 4);
  float*  partQ  = (float*)take((size_t)NB * CCH * SBLK * 4);
  float*  sb     = (float*)take((size_t)2 * CCH * 4);

  k_theta<<<(NB * S_TOT + 255) / 256, 256, 0, stream>>>(x, thetaB);
  k_phi  <<<(NB * T_PAD + 255) / 256, 256, 0, stream>>>(x, phiT);
  k_g    <<<dim3((T_PAD + 255) / 256, ACH, NB), 256, 0, stream>>>(x, g_w, g_b, gxT);
  k_attn <<<(NB * ROWBLK) / 4, 128, 0, stream>>>(thetaB, phiT, gxT, y);
  k_z    <<<dim3(SBLK, CCH, NB), 256, 0, stream>>>(y, w_w, w_b, z, partS, partQ);
  k_bn   <<<1, CCH, 0, stream>>>(partS, partQ, gamma, beta, sb);
  k_out  <<<(int)(((size_t)NB * CCH * S_TOT + 255) / 256), 256, 0, stream>>>(z, sb, x, out);
}
